// EgoAttention_5025111736702
// MI455X (gfx1250) — compile-verified, matrix operand order checked by reading, not run on hardware
//
#include <hip/hip_runtime.h>
#include <cstdint>

typedef __attribute__((ext_vector_type(16))) __bf16 v16bf;
typedef __attribute__((ext_vector_type(8)))  __bf16 v8bf;
typedef __attribute__((ext_vector_type(8)))  float  v8f;
typedef __attribute__((ext_vector_type(4)))  float  v4f;

// Problem constants
#define BA   4096   // B*A
#define Cc   128
#define Nn   64
#define Hh   8
#define Dd   16

// ---------------------------------------------------------------------------
// Pre-pack Wq/Wk/Wv (f32, row-major 128x128) into bf16 WMMA B-fragment order.
// Tile index = mat*32 + n_tile*4 + k_tile. Per tile: 32 lanes x 16 halfs.
// Element i of lane l:  col n = n_tile*16 + (l&15)
//                       k     = k_tile*32 + (i&7) + ((i>>3)<<4) + ((l>>4)<<3)
// ---------------------------------------------------------------------------
__global__ __launch_bounds__(32)
void pack_weights_kernel(const float* __restrict__ Wq,
                         const float* __restrict__ Wk,
                         const float* __restrict__ Wv,
                         __bf16* __restrict__ packed) {
    const int blk    = blockIdx.x;        // 0..95
    const int mat    = blk >> 5;          // 0=Wq 1=Wk 2=Wv
    const int tile   = blk & 31;
    const int n_tile = tile >> 2;
    const int k_tile = tile & 3;
    const int l      = threadIdx.x;       // 0..31
    const float* src = (mat == 0) ? Wq : ((mat == 1) ? Wk : Wv);
    const int n  = n_tile * 16 + (l & 15);
    const int kb = k_tile * 32 + ((l >> 4) << 3);
    __bf16* dst = packed + ((size_t)blk * 32 + l) * 16;
#pragma unroll
    for (int i = 0; i < 16; ++i) {
        int k = kb + (i & 7) + ((i >> 3) << 4);
        dst[i] = (__bf16)src[k * Cc + n];
    }
}

// ---------------------------------------------------------------------------
// Fused ego-attention: one workgroup (8 waves) per (b,a).
// ---------------------------------------------------------------------------
__global__ __launch_bounds__(256, 2)
void ego_attention_kernel(const float* __restrict__ x,
                          const float* __restrict__ y,
                          const float* __restrict__ t,
                          const float* __restrict__ W,
                          const float* __restrict__ pmask,
                          const float* __restrict__ cmask,
                          const __bf16* __restrict__ wpack,
                          const float* __restrict__ Wo,
                          const float* __restrict__ ln_g,
                          const float* __restrict__ ln_b,
                          float* __restrict__ out) {
    __shared__ __bf16 k_bf[Nn * Cc];     // 16 KB  LayerNormed k rows (bf16)
    __shared__ __bf16 q_bf[16 * Cc];     //  4 KB  q in row 0, rows 1..15 zero
    __shared__ __bf16 kh_bf[Nn * Cc];    // 16 KB  k @ Wk
    __shared__ __bf16 vh_bf[Nn * Cc];    // 16 KB  k @ Wv
    __shared__ float  qh[Cc];            //  q @ Wq
    __shared__ float  mvec[Cc];          //  attention output (H*D)
    __shared__ float  att_w[Hh * Nn];    //  att weights / reused as partials

    const int tid  = threadIdx.x;
    const int wave = tid >> 5;
    const int lane = tid & 31;
    const int ba   = blockIdx.x;                 // b*A + a
    const size_t rowbase = (size_t)ba * Nn;      // row in (B*A*N, C)

    // zero q tile (rows 1..15 must not contaminate via NaN garbage)
    for (int i = tid; i < 16 * Cc / 2; i += 256) ((uint32_t*)q_bf)[i] = 0u;

    // ------------------ Stage A: fused scale-add + LayerNorm ------------------
    const int c0 = lane * 4;                     // 4 channels per lane
    const v4f tv = *(const v4f*)(t    + c0);
    const v4f gv = *(const v4f*)(ln_g + c0);
    const v4f bv = *(const v4f*)(ln_b + c0);

    for (int r = wave; r < Nn; r += 8) {
        const float* yrow = y + (rowbase + r) * Cc;
        const float* Wrow = W + (rowbase + r) * Cc;
        v4f z = *(const v4f*)(yrow + c0) * *(const v4f*)(Wrow + c0) + tv;
        float s1 = z[0] + z[1] + z[2] + z[3];
        float s2 = z[0]*z[0] + z[1]*z[1] + z[2]*z[2] + z[3]*z[3];
#pragma unroll
        for (int off = 1; off < 32; off <<= 1) {
            s1 += __shfl_xor(s1, off, 32);
            s2 += __shfl_xor(s2, off, 32);
        }
        const float mu  = s1 * (1.0f / Cc);
        const float var = s2 * (1.0f / Cc) - mu * mu;
        const float inv = rsqrtf(var + 1e-5f);
#pragma unroll
        for (int j = 0; j < 4; ++j)
            k_bf[r * Cc + c0 + j] = (__bf16)((z[j] - mu) * inv * gv[j] + bv[j]);
    }
    if (wave == 0) {                              // q row: x * W[...,0,:] + t
        const float* xrow = x + (size_t)ba * Cc;
        const float* Wrow = W + rowbase * Cc;
        v4f z = *(const v4f*)(xrow + c0) * *(const v4f*)(Wrow + c0) + tv;
        float s1 = z[0] + z[1] + z[2] + z[3];
        float s2 = z[0]*z[0] + z[1]*z[1] + z[2]*z[2] + z[3]*z[3];
#pragma unroll
        for (int off = 1; off < 32; off <<= 1) {
            s1 += __shfl_xor(s1, off, 32);
            s2 += __shfl_xor(s2, off, 32);
        }
        const float mu  = s1 * (1.0f / Cc);
        const float var = s2 * (1.0f / Cc) - mu * mu;
        const float inv = rsqrtf(var + 1e-5f);
#pragma unroll
        for (int j = 0; j < 4; ++j)
            q_bf[c0 + j] = (__bf16)((z[j] - mu) * inv * gv[j] + bv[j]);
    }
    __syncthreads();

    // ------------------ Stage B: WMMA GEMMs (bf16 in, f32 acc) ----------------
    // wave w owns output column tile w (cols 16w..16w+15)
    v8f acck[4] = {};
    v8f accv[4] = {};
    v8f accq   = {};

    const __bf16* wq_p = wpack + ((size_t)(0 * 32 + wave * 4) * 32 + lane) * 16;
    const __bf16* wk_p = wpack + ((size_t)(1 * 32 + wave * 4) * 32 + lane) * 16;
    const __bf16* wv_p = wpack + ((size_t)(2 * 32 + wave * 4) * 32 + lane) * 16;

    const int arow   = lane & 15;             // A-fragment row within tile
    const int klane  = (lane >> 4) << 3;      // +8 for upper half-wave

#pragma unroll
    for (int kt = 0; kt < 4; ++kt) {
        union F16 { v16bf v; v8bf h[2]; };
        F16 bq, bk, bv2;
        bq.v  = *(const v16bf*)(wq_p + kt * 512);
        bk.v  = *(const v16bf*)(wk_p + kt * 512);
        bv2.v = *(const v16bf*)(wv_p + kt * 512);
        const int kbase = kt * 32 + klane;
#pragma unroll
        for (int mt = 0; mt < 4; ++mt) {
            F16 a;
            const __bf16* ap = &k_bf[(mt * 16 + arow) * Cc + kbase];
            a.h[0] = *(const v8bf*)(ap);
            a.h[1] = *(const v8bf*)(ap + 16);
            acck[mt] = __builtin_amdgcn_wmma_f32_16x16x32_bf16(
                false, a.v, false, bk.v,  (short)0, acck[mt], false, false);
            accv[mt] = __builtin_amdgcn_wmma_f32_16x16x32_bf16(
                false, a.v, false, bv2.v, (short)0, accv[mt], false, false);
        }
        {
            F16 aq;
            const __bf16* ap = &q_bf[arow * Cc + kbase];
            aq.h[0] = *(const v8bf*)(ap);
            aq.h[1] = *(const v8bf*)(ap + 16);
            accq = __builtin_amdgcn_wmma_f32_16x16x32_bf16(
                false, aq.v, false, bq.v, (short)0, accq, false, false);
        }
    }

    // C/D layout: lane l -> col (l&15), VGPR j -> row ((l>>4)*8 + j)
    const int dcol = wave * 16 + (lane & 15);
#pragma unroll
    for (int mt = 0; mt < 4; ++mt) {
#pragma unroll
        for (int j = 0; j < 8; ++j) {
            const int row = mt * 16 + ((lane >> 4) << 3) + j;
            kh_bf[row * Cc + dcol] = (__bf16)acck[mt][j];
            vh_bf[row * Cc + dcol] = (__bf16)accv[mt][j];
        }
    }
    if (lane < 16) qh[wave * 16 + lane] = accq[0];   // only q row 0 is real
    __syncthreads();

    // ------------------ Stage C: attention (wave = head) ----------------------
    const int h = wave;
    float lg0 = 0.0f, lg1 = 0.0f;                // n0 = lane, n1 = lane+32
#pragma unroll
    for (int d = 0; d < Dd; ++d) {
        const float qv = qh[h * Dd + d];
        lg0 += qv * (float)kh_bf[lane * Cc + h * Dd + d];
        lg1 += qv * (float)kh_bf[(lane + 32) * Cc + h * Dd + d];
    }
    lg0 *= 0.25f;  lg1 *= 0.25f;                 // 1/sqrt(D)

    float mx = fmaxf(lg0, lg1);
#pragma unroll
    for (int off = 1; off < 32; off <<= 1) mx = fmaxf(mx, __shfl_xor(mx, off, 32));
    float e0 = __expf(lg0 - mx), e1 = __expf(lg1 - mx);
    float s = e0 + e1;
#pragma unroll
    for (int off = 1; off < 32; off <<= 1) s += __shfl_xor(s, off, 32);

    const float* pmr = pmask + (size_t)ba * Nn;
    const float* cmr = cmask + (size_t)ba * Nn;
    const float p0 = pmr[lane], p1 = pmr[lane + 32];
    float a0 = (e0 / s) * p0 * cmr[lane];
    float a1 = (e1 / s) * p1 * cmr[lane + 32];
    float s2r = a0 + a1;
#pragma unroll
    for (int off = 1; off < 32; off <<= 1) s2r += __shfl_xor(s2r, off, 32);
    const float inv2 = 1.0f / s2r;
    att_w[h * Nn + lane]      = a0 * inv2 * p0;   // reference applies pm again
    att_w[h * Nn + lane + 32] = a1 * inv2 * p1;
    __syncthreads();

    // m[h][d] = sum_n att_w[h][n] * vh[n][h*16+d]; lanes split n in halves
    {
        const int d  = lane & 15;
        const int nb = (lane >> 4) * 32;
        float acc = 0.0f;
#pragma unroll
        for (int i = 0; i < 32; ++i)
            acc += att_w[h * Nn + nb + i] * (float)vh_bf[(nb + i) * Cc + h * Dd + d];
        acc += __shfl_xor(acc, 16, 32);
        if (lane < 16) mvec[h * Dd + d] = acc;
    }
    __syncthreads();

    // ------------------ Stage D: out = mvec @ Wo ------------------------------
    {
        const int j    = tid & 127;
        const int half = tid >> 7;
        float acc = 0.0f;
#pragma unroll 8
        for (int kk = half * 64; kk < half * 64 + 64; ++kk)
            acc += mvec[kk] * Wo[kk * Cc + j];
        att_w[tid] = acc;                        // reuse att_w as 256 partials
        __syncthreads();
        if (tid < 128)
            out[(size_t)ba * Cc + tid] = att_w[tid] + att_w[tid + 128];
    }
}

// ---------------------------------------------------------------------------
extern "C" void kernel_launch(void* const* d_in, const int* in_sizes, int n_in,
                              void* d_out, int out_size, void* d_ws, size_t ws_size,
                              hipStream_t stream) {
    // setup_inputs order: e, x, y, t, W, pairwise_mask, cutoff_mask,
    //                     Wq, Wk, Wv, Wo, ln_g, ln_b   ('e' is unused)
    const float* x    = (const float*)d_in[1];
    const float* y    = (const float*)d_in[2];
    const float* t    = (const float*)d_in[3];
    const float* W    = (const float*)d_in[4];
    const float* pm   = (const float*)d_in[5];
    const float* cm   = (const float*)d_in[6];
    const float* Wq   = (const float*)d_in[7];
    const float* Wk   = (const float*)d_in[8];
    const float* Wv   = (const float*)d_in[9];
    const float* Wo   = (const float*)d_in[10];
    const float* ln_g = (const float*)d_in[11];
    const float* ln_b = (const float*)d_in[12];

    __bf16* wpack = (__bf16*)d_ws;   // 3 * 32 tiles * 32 lanes * 16 halfs = 96 KB

    pack_weights_kernel<<<96, 32, 0, stream>>>(Wq, Wk, Wv, wpack);
    ego_attention_kernel<<<BA, 256, 0, stream>>>(x, y, t, W, pm, cm, wpack, Wo,
                                                 ln_g, ln_b, (float*)d_out);
}
